// ToeplitzLinear_35399120454032
// MI455X (gfx1250) — matrix operand order repaired and checked by reading, not compiled, not run on hardware
//
#include <hip/hip_runtime.h>

// ---------------------------------------------------------------------------
// ToeplitzLinear: out[b,i] = sum_j x[b,j] * d[i+j]
//   == GEMM  C[8192x4096] = X[8192x4096] * B[4096x4096],  B[k,n] = d[k+n]
// Compute-bound (AI ~ 1070 FLOP/B). Strategy:
//   * bf16 hi/lo split of both operands, 3x v_wmma_f32_16x16x32_bf16 per
//     k-step (hi*hi + hi*lo + lo*hi), f32 accumulation -> ~fp32 accuracy at
//     3/8 the cost of the f32-WMMA (16x16x4) path.
//   * B (Hankel) never touches HBM: each block only needs a 4223-entry
//     window of diagonals, pre-split to bf16 hi/lo in LDS; a B fragment is
//     16 consecutive entries per lane (contiguous k = 16*half + e, A matches).
//   * X staged through a 1-deep register pipeline: next chunk's
//     global_load_b128s are issued before the current chunk's 48 WMMAs.
//   * 64x32 wave tile (8 accumulators): 24 WMMAs per 32-wide k-step per
//     wave vs ~40 ds_load_b128 -> matrix pipe is the bottleneck, as it
//     should be on a compute-bound kernel.
// ---------------------------------------------------------------------------

typedef __attribute__((ext_vector_type(16))) __bf16 v16bf;
typedef __attribute__((ext_vector_type(8)))  __bf16 v8bf;
typedef __attribute__((ext_vector_type(8)))  float  v8f;

constexpr int IN_F  = 4096;
constexpr int OUT_F = 4096;
constexpr int BATCH = 8192;
constexpr int NDIAG = IN_F + OUT_F - 1;     // 8191

constexpr int M_BLK = 128;                  // block tile rows (batch)
constexpr int N_BLK = 128;                  // block tile cols (out features)
constexpr int KC    = 64;                   // K chunk staged in LDS
constexpr int XSTR  = KC + 8;               // LDS row stride (keeps 16B align)
constexpr int DWIN  = N_BLK + IN_F - 1;     // 4223 diagonal entries per block
constexpr int DLDS  = 4288;                 // padded window

__global__ __launch_bounds__(256)
void toeplitz_wmma_kernel(const float* __restrict__ x,
                          const float* __restrict__ diag,
                          float* __restrict__ out)
{
    __shared__ __bf16 sD_hi[DLDS];
    __shared__ __bf16 sD_lo[DLDS];
    __shared__ __bf16 sX_hi[M_BLK * XSTR];
    __shared__ __bf16 sX_lo[M_BLK * XSTR];

    const int tid  = threadIdx.x;
    const int wave = tid >> 5;           // 8 waves (wave32)
    const int lane = tid & 31;
    const int half = lane >> 4;          // half-wave (ISA frag layout)
    const int r16  = lane & 15;
    const int wm   = wave >> 2;          // 0..1 : M position (64 rows each)
    const int wn   = wave & 3;           // 0..3 : N position (32 cols each)

    const int mBase = blockIdx.y * M_BLK;
    const int nBase = blockIdx.x * N_BLK;

    // ---- stage this block's diagonal window, split into bf16 hi/lo -------
    for (int i = tid; i < DLDS; i += 256) {
        int   g = nBase + i;
        float v = (i < DWIN && g < NDIAG) ? diag[g] : 0.0f;
        __bf16 h = (__bf16)v;
        sD_hi[i] = h;
        sD_lo[i] = (__bf16)(v - (float)h);
    }

    // ---- X staging: each thread owns 4 runs of 8 consecutive K elems -----
    const int r32  = tid >> 3;           // 0..31 : row within a 32-row pass
    const int col8 = (tid & 7) * 8;      // 0..56 : 8-wide K run

    float4 rA[4], rB[4];                 // register pipeline (next chunk)
    auto fetch = [&](int k0) {
        #pragma unroll
        for (int g = 0; g < 4; ++g) {
            const int row = r32 + g * 32;
            const float* p = x + (size_t)(mBase + row) * IN_F + k0 + col8;
            rA[g] = *(const float4*)(p);
            rB[g] = *(const float4*)(p + 4);
        }
    };
    auto stage = [&]() {                 // regs -> LDS, bf16 hi/lo, b128 stores
        #pragma unroll
        for (int g = 0; g < 4; ++g) {
            const int row = r32 + g * 32;
            const float f[8] = {rA[g].x, rA[g].y, rA[g].z, rA[g].w,
                                rB[g].x, rB[g].y, rB[g].z, rB[g].w};
            v8bf h8, l8;
            #pragma unroll
            for (int j = 0; j < 8; ++j) {
                __bf16 h = (__bf16)f[j];
                h8[j] = h;
                l8[j] = (__bf16)(f[j] - (float)h);
            }
            *(v8bf*)(sX_hi + row * XSTR + col8) = h8;
            *(v8bf*)(sX_lo + row * XSTR + col8) = l8;
        }
    };

    v8f acc[8];                          // [mi][ni] -> acc[mi*2+ni]
    #pragma unroll
    for (int i = 0; i < 8; ++i) acc[i] = {};

    // ---- prologue: chunk 0 through the pipeline --------------------------
    fetch(0);
    stage();
    __syncthreads();                     // diag + chunk 0 visible

    for (int k0 = 0; k0 < IN_F; k0 += KC) {
        const bool more = (k0 + KC) < IN_F;
        if (more) fetch(k0 + KC);        // global loads in flight during WMMAs

        #pragma unroll
        for (int kk = 0; kk < KC; kk += 32) {
            // ---- B fragments (Hankel) straight from the diagonal table ---
            // B layout used here (contiguous k): lane%16 = N, lanes>=16 carry K+16, element e = K.
            // value = d[(nBase + N) + (k + K)] -> 16 consecutive entries.
            v16bf bh[2] = {}, bl[2] = {};
            #pragma unroll
            for (int ni = 0; ni < 2; ++ni) {
                const int base = wn * 32 + ni * 16 + r16 + k0 + kk + half * 16;
                #pragma unroll
                for (int e = 0; e < 16; ++e) {
                    bh[ni][e] = sD_hi[base + e];
                    bl[ni][e] = sD_lo[base + e];
                }
            }
            // ---- A fragments: 4 x (16x32) row tiles from staged X --------
            // A layout: lane%16 = M; lanes<16 K run {0..15},
            // lanes>=16 K run {16..31} (matches B).
            const int koff = kk + half * 16;
            #pragma unroll
            for (int mi = 0; mi < 4; ++mi) {
                const int row = wm * 64 + mi * 16 + r16;
                const v8bf h0 = *(const v8bf*)(sX_hi + row * XSTR + koff);
                const v8bf h1 = *(const v8bf*)(sX_hi + row * XSTR + koff + 8);
                const v8bf l0 = *(const v8bf*)(sX_lo + row * XSTR + koff);
                const v8bf l1 = *(const v8bf*)(sX_lo + row * XSTR + koff + 8);
                const v16bf ah = __builtin_shufflevector(h0, h1,
                        0,1,2,3,4,5,6,7,8,9,10,11,12,13,14,15);
                const v16bf al = __builtin_shufflevector(l0, l1,
                        0,1,2,3,4,5,6,7,8,9,10,11,12,13,14,15);
                // ---- 3-term bf16 split accumulation, f32 accumulators ----
                #pragma unroll
                for (int ni = 0; ni < 2; ++ni) {
                    v8f a = acc[mi * 2 + ni];
                    a = __builtin_amdgcn_wmma_f32_16x16x32_bf16(
                        false, ah, false, bh[ni], (short)0, a, false, false);
                    a = __builtin_amdgcn_wmma_f32_16x16x32_bf16(
                        false, ah, false, bl[ni], (short)0, a, false, false);
                    a = __builtin_amdgcn_wmma_f32_16x16x32_bf16(
                        false, al, false, bh[ni], (short)0, a, false, false);
                    acc[mi * 2 + ni] = a;
                }
            }
        }

        __syncthreads();                 // all waves done reading this chunk
        if (more) stage();               // write next chunk (regs ready)
        __syncthreads();                 // next chunk visible
    }

    // ---- epilogue: C/D layout is VGPR r -> M = r + 8*half, lane%16 = N ---
    #pragma unroll
    for (int mi = 0; mi < 4; ++mi) {
        const int row0 = mBase + wm * 64 + mi * 16 + half * 8;
        #pragma unroll
        for (int ni = 0; ni < 2; ++ni) {
            const int col = nBase + wn * 32 + ni * 16 + r16;
            #pragma unroll
            for (int r = 0; r < 8; ++r)
                out[(size_t)(row0 + r) * OUT_F + col] = acc[mi * 2 + ni][r];
        }
    }
}

extern "C" void kernel_launch(void* const* d_in, const int* in_sizes, int n_in,
                              void* d_out, int out_size, void* d_ws, size_t ws_size,
                              hipStream_t stream) {
    const float* x    = (const float*)d_in[0];
    const float* diag = (const float*)d_in[1];
    float*       out  = (float*)d_out;

    dim3 grid(OUT_F / N_BLK, BATCH / M_BLK);   // (32, 64)
    toeplitz_wmma_kernel<<<grid, dim3(256), 0, stream>>>(x, diag, out);
}
